// Framework_29291676958728
// MI455X (gfx1250) — matrix operand order repaired and checked by reading, not compiled, not run on hardware
//
#include <hip/hip_runtime.h>
#include <hip/hip_bf16.h>
#include <math.h>

typedef __bf16 v16bf __attribute__((ext_vector_type(16)));
typedef float  v8f   __attribute__((ext_vector_type(8)));

#define Bn 4
#define Hn 256
#define Wn 256
#define C1n 256
#define C2n 128
#define CFn 384          // C1+C2
#define Kc 128           // cells per patch
#define Nn (Bn*Kc)       // 512 cells total
#define En 256
#define NCn 16
#define NCpad 32         // zero-padded K for the q projection
#define Gn 512
#define Rn 64
#define HEADSn 8
#define DHn 32
#define HW (Hn*Wn)

// ---------------------------------------------------------------------------
// Stage A: stream feat once; per-(b,c) block accumulates K=128 cell sums in
// LDS (ds_add_f32), block-reduces the global sum, c==0 blocks also count
// cell areas. Every global destination is written by exactly one block.
// ---------------------------------------------------------------------------
__global__ void seg_stream(const float* __restrict__ hd1,
                           const float* __restrict__ h1,
                           const int*   __restrict__ mask,
                           float* __restrict__ sums,   // [512,384]
                           float* __restrict__ glob,   // [4,384] (raw sums)
                           float* __restrict__ area)   // [512] (raw counts)
{
    int bc = blockIdx.x;            // 0 .. B*CF-1
    int b  = bc / CFn;
    int c  = bc - b * CFn;
    const float* src = (c < C1n) ? (hd1 + ((size_t)b * C1n + c) * HW)
                                 : (h1  + ((size_t)b * C2n + (c - C1n)) * HW);
    const int* m = mask + (size_t)b * HW;

    __shared__ float ls[Kc];
    __shared__ float lcnt[Kc];
    __shared__ float red[256];
    int t = threadIdx.x;
    if (t < Kc) { ls[t] = 0.f; lcnt[t] = 0.f; }
    __syncthreads();

    float g = 0.f;
    for (int p = t; p < HW; p += 256) {
        __builtin_prefetch(src + p + 4096, 0, 0);   // global_prefetch_b8
        float v = src[p];
        int   id = m[p];
        g += v;
        if (id > 0) {
            atomicAdd(&ls[id - 1], v);              // ds_add_f32
            if (c == 0) atomicAdd(&lcnt[id - 1], 1.0f);
        }
    }
    red[t] = g;
    __syncthreads();
    for (int s = 128; s > 0; s >>= 1) {
        if (t < s) red[t] += red[t + s];
        __syncthreads();
    }
    if (t == 0) glob[b * CFn + c] = red[0];
    if (t < Kc) {
        sums[((size_t)b * Kc + t) * CFn + c] = ls[t];
        if (c == 0) area[b * Kc + t] = lcnt[t];
    }
}

// all_fv[n, 0:384] = sums[n]/max(area,1) ; all_fv[n, 384:768] = glob[b]/HW
__global__ void finalize_fv(const float* __restrict__ sums,
                            const float* __restrict__ glob,
                            const float* __restrict__ area,
                            float* __restrict__ all_fv)
{
    int idx = blockIdx.x * blockDim.x + threadIdx.x;
    if (idx >= Nn * 2 * CFn) return;
    int n = idx / (2 * CFn);
    int c = idx - n * (2 * CFn);
    float v;
    if (c < CFn) v = sums[n * CFn + c] / fmaxf(area[n], 1.0f);
    else         v = glob[(n >> 7) * CFn + (c - CFn)] * (1.0f / (float)HW);
    all_fv[idx] = v;
}

// build zero-padded Wq: wqpad[3][32][256]; rows 16..31 = 0
__global__ void pad_wq(const float* __restrict__ wq, float* __restrict__ wqpad)
{
    int idx = blockIdx.x * blockDim.x + threadIdx.x;
    if (idx >= 3 * NCpad * En) return;
    int col = idx & (En - 1);
    int row = (idx >> 8) & (NCpad - 1);
    int i   = idx >> 13;
    wqpad[idx] = (row < NCn) ? wq[(i * NCn + row) * En + col] : 0.f;
}

// ---------------------------------------------------------------------------
// Generic bf16 WMMA GEMM: D = act(scale * (A @ B[^T]) [+ Cadd])
// CONTRACT: M % 16 == 0, N % 16 == 0, Kd % 32 == 0 (no bounds checks).
// One wave per 16x16 output tile; K swept in steps of 32 via
// v_wmma_f32_16x16x32_bf16 (f32 accumulate); f32 -> bf16 in registers.
// TRANSB is a template parameter so each instantiation has a branch-free
// inner loop; A/B pointers are strength-reduced to per-step increments.
// ---------------------------------------------------------------------------
template <int TRANSB>
__global__ void gemm_bf16_wmma(const float* __restrict__ A,
                               const float* __restrict__ B,
                               const float* __restrict__ Cadd,
                               float* __restrict__ D,
                               int M, int N, int Kd,
                               int lda, int ldb, int ldd, int ldc,
                               int aOff, int bOff, int dOff,
                               int doRelu, float scale)
{
    int gwave = (blockIdx.x * blockDim.x + threadIdx.x) >> 5;
    int lane  = threadIdx.x & 31;
    int tilesN = N >> 4;
    int tiles  = (M >> 4) * tilesN;
    if (gwave >= tiles) return;
    int tm = gwave / tilesN, tn = gwave - tm * tilesN;
    int m0 = tm << 4, n0 = tn << 4;
    int r  = lane & 15;     // A-tile row / B-tile col owned by this lane
    int hi = lane >> 4;

    v8f acc = {0.f, 0.f, 0.f, 0.f, 0.f, 0.f, 0.f, 0.f};

    const float* aPtr = A + aOff + (size_t)(m0 + r) * lda;
    const float* bPtr = TRANSB
        ? (B + bOff + (size_t)(n0 + r) * ldb + hi * 16)
        : (B + bOff + (size_t)(hi * 16) * ldb + n0 + r);
    const size_t bStep = TRANSB ? 32 : (size_t)32 * ldb;

#pragma unroll 2
    for (int kk = 0; kk < Kd; kk += 32) {
        v16bf av, bv;
        // A: VGPR v holds K = hi*16 + 2v , +1  (aligned float2)
#pragma unroll
        for (int v = 0; v < 8; ++v) {
            int ka = hi * 16 + 2 * v;
            float2 a2 = *(const float2*)(aPtr + ka);
            av[2 * v]     = (__bf16)a2.x;
            av[2 * v + 1] = (__bf16)a2.y;
        }
        // B: lane col = r, VGPR v holds K = hi*16 + 2v , +1
        if (TRANSB) {
#pragma unroll
            for (int v = 0; v < 8; ++v) {
                float2 b2 = *(const float2*)(bPtr + 2 * v);
                bv[2 * v]     = (__bf16)b2.x;
                bv[2 * v + 1] = (__bf16)b2.y;
            }
        } else {
#pragma unroll
            for (int v = 0; v < 8; ++v) {
                bv[2 * v]     = (__bf16)bPtr[(size_t)(2 * v) * ldb];
                bv[2 * v + 1] = (__bf16)bPtr[(size_t)(2 * v + 1) * ldb];
            }
        }
        acc = __builtin_amdgcn_wmma_f32_16x16x32_bf16(
            false, av, false, bv, (short)0, acc, false, false);
        aPtr += 32;
        bPtr += bStep;
    }

#pragma unroll
    for (int v = 0; v < 8; ++v) {
        int mrow = m0 + v + 8 * hi;
        int ncol = n0 + r;
        float rres = acc[v] * scale;
        if (Cadd) rres += Cadd[(size_t)mrow * ldc + ncol];
        if (doRelu) rres = fmaxf(rres, 0.f);
        D[dOff + (size_t)mrow * ldd + ncol] = rres;
    }
}

// ---------------------------------------------------------------------------
// Row softmax in place, one block (256 threads) per row.
// ---------------------------------------------------------------------------
__global__ void row_softmax(float* __restrict__ X, int cols)
{
    float* x = X + (size_t)blockIdx.x * cols;
    __shared__ float red[256];
    int t = threadIdx.x;
    float mx = -3.4e38f;
    for (int c = t; c < cols; c += 256) mx = fmaxf(mx, x[c]);
    red[t] = mx; __syncthreads();
    for (int s = 128; s > 0; s >>= 1) { if (t < s) red[t] = fmaxf(red[t], red[t + s]); __syncthreads(); }
    mx = red[0]; __syncthreads();
    float sum = 0.f;
    for (int c = t; c < cols; c += 256) { float e = __expf(x[c] - mx); x[c] = e; sum += e; }
    red[t] = sum; __syncthreads();
    for (int s = 128; s > 0; s >>= 1) { if (t < s) red[t] += red[t + s]; __syncthreads(); }
    float inv = 1.0f / red[0];
    for (int c = t; c < cols; c += 256) x[c] *= inv;
}

// emb_patches[b,c] = mean over 128 cells of emb
__global__ void mean_emb(const float* __restrict__ emb, float* __restrict__ embp)
{
    int b = blockIdx.x, c = threadIdx.x;   // 4 blocks x 256 threads
    float s = 0.f;
    for (int i = 0; i < Kc; ++i) s += emb[((size_t)(b * Kc + i)) * En + c];
    embp[b * En + c] = s * (1.0f / (float)Kc);
}

// comp = softmax(relu(embp @ W1) @ W2); also emit comp_tiled [512,32] (zero-pad)
__global__ void comp_kernel(const float* __restrict__ embp,
                            const float* __restrict__ W1,
                            const float* __restrict__ W2,
                            float* __restrict__ comp_out,
                            float* __restrict__ comp_tiled)
{
    __shared__ float h[Bn * En];
    __shared__ float cs[Bn * NCn];
    int t = threadIdx.x;
    for (int idx = t; idx < Bn * En; idx += 256) {
        int b = idx >> 8, c = idx & 255;
        float s = 0.f;
        for (int k = 0; k < En; ++k) s += embp[b * En + k] * W1[k * En + c];
        h[idx] = fmaxf(s, 0.f);
    }
    __syncthreads();
    for (int idx = t; idx < Bn * NCn; idx += 256) {
        int b = idx >> 4, j = idx & 15;
        float s = 0.f;
        for (int k = 0; k < En; ++k) s += h[b * En + k] * W2[k * NCn + j];
        cs[idx] = s;
    }
    __syncthreads();
    if (t < Bn) {
        float mx = -3.4e38f;
        for (int j = 0; j < NCn; ++j) mx = fmaxf(mx, cs[t * NCn + j]);
        float sum = 0.f;
        for (int j = 0; j < NCn; ++j) { float e = __expf(cs[t * NCn + j] - mx); cs[t * NCn + j] = e; sum += e; }
        float inv = 1.0f / sum;
        for (int j = 0; j < NCn; ++j) cs[t * NCn + j] *= inv;
    }
    __syncthreads();
    for (int idx = t; idx < Bn * NCn; idx += 256) comp_out[idx] = cs[idx];
    for (int idx = t; idx < Nn * NCpad; idx += 256) {
        int n = idx >> 5, j = idx & 31;
        comp_tiled[idx] = (j < NCn) ? cs[(n >> 7) * NCn + j] : 0.f;
    }
}

// ---------------------------------------------------------------------------
static inline void launch_gemm(hipStream_t s, const float* A, const float* B,
                               const float* Cadd, float* D,
                               int M, int N, int K, int lda, int ldb, int ldd, int ldc,
                               int aOff, int bOff, int dOff,
                               int transB, int relu, float scale)
{
    int tiles = (M / 16) * (N / 16);
    int blocks = (tiles + 7) / 8;     // 8 waves (256 threads) per block
    if (transB)
        gemm_bf16_wmma<1><<<blocks, 256, 0, s>>>(A, B, Cadd, D, M, N, K, lda, ldb, ldd,
                                                 ldc, aOff, bOff, dOff, relu, scale);
    else
        gemm_bf16_wmma<0><<<blocks, 256, 0, s>>>(A, B, Cadd, D, M, N, K, lda, ldb, ldd,
                                                 ldc, aOff, bOff, dOff, relu, scale);
}

extern "C" void kernel_launch(void* const* d_in, const int* in_sizes, int n_in,
                              void* d_out, int out_size, void* d_ws, size_t ws_size,
                              hipStream_t stream)
{
    (void)in_sizes; (void)n_in; (void)out_size; (void)ws_size;
    const float* hd1      = (const float*)d_in[0];
    const float* h1       = (const float*)d_in[1];
    const float* ref_orig = (const float*)d_in[2];   // [64,512]
    const float* embed_w  = (const float*)d_in[3];   // [768,256]
    const float* comp_w1  = (const float*)d_in[4];
    const float* comp_w2  = (const float*)d_in[5];
    const float* hist_w1  = (const float*)d_in[6];
    const float* hist_w2  = (const float*)d_in[7];
    const float* genes_w1 = (const float*)d_in[8];   // [512,256]
    const float* genes_w2 = (const float*)d_in[9];   // [256,16]
    const float* wref_w1  = (const float*)d_in[10];  // [3,256,256]
    const float* wref_w2  = (const float*)d_in[11];  // [3,256,64]
    const float* ca_wq    = (const float*)d_in[12];  // [3,16,256]
    const float* ca_wk    = (const float*)d_in[13];  // [3,512,256]
    const float* ca_wv    = (const float*)d_in[14];  // [3,512,256]
    const float* ca_wo    = (const float*)d_in[15];  // [3,256,512]
    const int*   mask     = (const int*)d_in[16];
    float* out = (float*)d_out;

    // d_out layout (flat f32, reference return order)
    const int O_CT   = 0;                      // [512,16]
    const int O_EXP0 = O_CT + Nn * NCn;        // [512,512] x3
    const int O_CTE  = O_EXP0 + 3 * Nn * Gn;   // [512,16]
    const int O_GH   = O_CTE + Nn * NCn;       // [512,256]
    const int O_COMP = O_GH + Nn * En;         // [4,16]
    const int O_AREA = O_COMP + Bn * NCn;      // [512]

    // workspace layout (floats)
    float* ws = (float*)d_ws;
    float* sums    = ws;                       // 512*384
    float* glob    = sums + Nn * CFn;          // 4*384
    float* all_fv  = glob + Bn * CFn;          // 512*768
    float* emb     = all_fv + Nn * 2 * CFn;    // 512*256
    float* embp    = emb + Nn * En;            // 4*256
    float* ctile   = embp + Bn * En;           // 512*32 (zero-padded)
    float* wqpad   = ctile + Nn * NCpad;       // 3*32*256 (zero-padded)
    float* tmpA    = wqpad + 3 * NCpad * En;   // 512*256
    float* rw      = tmpA + Nn * En;           // 512*64
    float* refw    = rw + Nn * Rn;             // 512*512
    float* qbuf    = refw + Nn * Gn;           // 512*256
    float* kbuf    = qbuf + Nn * En;           // 512*256
    float* vbuf    = kbuf + Nn * En;           // 512*256
    float* attn    = vbuf + Nn * En;           // 512*512 (per-head reuse)
    float* obuf    = attn + Nn * Nn;           // 512*256

    // A: stream feat once; cell sums (LDS atomics), global sums, areas
    seg_stream<<<Bn * CFn, 256, 0, stream>>>(hd1, h1, mask, sums, glob, out + O_AREA);
    finalize_fv<<<(Nn * 2 * CFn + 255) / 256, 256, 0, stream>>>(sums, glob, out + O_AREA, all_fv);
    pad_wq<<<(3 * NCpad * En + 255) / 256, 256, 0, stream>>>(ca_wq, wqpad);

    // emb = all_fv @ embed_w   [512,768]x[768,256]
    launch_gemm(stream, all_fv, embed_w, nullptr, emb, Nn, En, 2 * CFn,
                2 * CFn, En, En, 0, 0, 0, 0, 0, 0, 1.0f);
    mean_emb<<<Bn, 256, 0, stream>>>(emb, embp);
    comp_kernel<<<1, 256, 0, stream>>>(embp, comp_w1, comp_w2, out + O_COMP, ctile);

    // out_cell_type = relu(emb @ hist_w1) @ hist_w2
    launch_gemm(stream, emb, hist_w1, nullptr, tmpA, Nn, En, En, En, En, En, 0, 0, 0, 0, 0, 1, 1.0f);
    launch_gemm(stream, tmpA, hist_w2, nullptr, out + O_CT, Nn, NCn, En, En, NCn, NCn, 0, 0, 0, 0, 0, 0, 1.0f);

    const float attnScale = 1.0f / sqrtf((float)DHn);
    for (int i = 0; i < 3; ++i) {
        float* expr = out + O_EXP0 + i * Nn * Gn;
        // rw = softmax(relu(emb @ wref_w1[i]) @ wref_w2[i])
        launch_gemm(stream, emb, wref_w1, nullptr, tmpA, Nn, En, En,
                    En, En, En, 0, 0, i * En * En, 0, 0, 1, 1.0f);
        launch_gemm(stream, tmpA, wref_w2, nullptr, rw, Nn, Rn, En,
                    En, Rn, Rn, 0, 0, i * En * Rn, 0, 0, 0, 1.0f);
        row_softmax<<<Nn, 256, 0, stream>>>(rw, Rn);
        // ref_weighted = rw @ ref_orig   [512,64]x[64,512]
        launch_gemm(stream, rw, ref_orig, nullptr, refw, Nn, Gn, Rn,
                    Rn, Gn, Gn, 0, 0, 0, 0, 0, 0, 1.0f);
        // q,k,v projections (q uses zero-padded K=32 operands)
        launch_gemm(stream, ctile, wqpad, nullptr, qbuf, Nn, En, NCpad,
                    NCpad, En, En, 0, 0, i * NCpad * En, 0, 0, 0, 1.0f);
        launch_gemm(stream, refw, ca_wk, nullptr, kbuf, Nn, En, Gn,
                    Gn, En, En, 0, 0, i * Gn * En, 0, 0, 0, 1.0f);
        launch_gemm(stream, refw, ca_wv, nullptr, vbuf, Nn, En, Gn,
                    Gn, En, En, 0, 0, i * Gn * En, 0, 0, 0, 1.0f);
        // per-head attention: scores = (Qh @ Kh^T)/sqrt(dh); softmax; o_h = A @ Vh
        for (int h = 0; h < HEADSn; ++h) {
            launch_gemm(stream, qbuf, kbuf, nullptr, attn, Nn, Nn, DHn,
                        En, En, Nn, 0, h * DHn, h * DHn, 0, 1, 0, attnScale);
            row_softmax<<<Nn, 256, 0, stream>>>(attn, Nn);
            launch_gemm(stream, attn, vbuf, nullptr, obuf, Nn, DHn, Nn,
                        Nn, En, En, 0, 0, h * DHn, h * DHn, 0, 0, 1.0f);
        }
        // out_exprs[i] = relu(refw + o @ ca_wo[i])
        launch_gemm(stream, obuf, ca_wo, refw, expr, Nn, Gn, En,
                    En, Gn, Gn, Gn, 0, i * En * Gn, 0, 0, 1, 1.0f);
    }

    // genes_h = relu(out_exprs[0] @ genes_w1); out_cell_type_expr = genes_h @ genes_w2
    launch_gemm(stream, out + O_EXP0, genes_w1, nullptr, out + O_GH, Nn, En, Gn,
                Gn, En, En, 0, 0, 0, 0, 0, 1, 1.0f);
    launch_gemm(stream, out + O_GH, genes_w2, nullptr, out + O_CTE, Nn, NCn, En,
                En, NCn, NCn, 0, 0, 0, 0, 0, 0, 1.0f);
}
